// GCN_54949811585434
// MI455X (gfx1250) — matrix operand order repaired and checked by reading, not compiled, not run on hardware
//
#include <hip/hip_runtime.h>

#define D      256
#define NROWS  100000
#define TM     64        // rows of agg per block in the GEMM
#define APITCH 132       // dwords per A row in LDS (= 264 bf16 = 528 B, conflict-avoiding pitch)
#define CHUNK  512       // edges per block in the SpMM

typedef __attribute__((ext_vector_type(16))) __bf16   v16bf;
typedef __attribute__((ext_vector_type(8)))  float    v8f;
typedef __attribute__((ext_vector_type(4)))  unsigned u32x4;
typedef __attribute__((ext_vector_type(2)))  float    f32x2;

static __device__ __forceinline__ unsigned pk_bf16(float a, float b) {
  union { __bf16 h[2]; unsigned u; } c;
  c.h[0] = (__bf16)a;
  c.h[1] = (__bf16)b;
  return c.u;
}

// ---------------------------------------------------------------------------
// Kernel A: pre-swizzle W (f32, [K=256][N=256] row-major) into bf16 B-fragment
// order. Element index = ((ks*16 + t)*32 + lane)*8 + v  where:
//   ks = K-step (32 K per step), t = 16-col N tile, lane = wave lane,
//   v  = B VGPR index; lane = h*16 + n, VGPR v holds K = ks*32 + h*16 + 2v, +1
// so each lane's 8 B dwords are contiguous (32 B -> two global_load_b128).
// Total 32768 dwords = 128 KB in d_ws (stays hot in L2).
// ---------------------------------------------------------------------------
__global__ __launch_bounds__(256) void gcn_wprep(const float* __restrict__ W,
                                                 unsigned* __restrict__ wsw) {
  int idx  = blockIdx.x * 256 + threadIdx.x;      // 0 .. 32767
  int v    = idx & 7;
  int lane = (idx >> 3) & 31;
  int t    = (idx >> 8) & 15;
  int ks   = idx >> 12;
  int n    = lane & 15;
  int h    = lane >> 4;
  int col  = t * 16 + n;
  int k    = ks * 32 + h * 16 + 2 * v;
  wsw[idx] = pk_bf16(W[k * D + col], W[(k + 1) * D + col]);
}

// ---------------------------------------------------------------------------
// Kernel B: SpMM. agg[r,:] += val * features[c,:]  (edge rows sorted).
// 256 threads = the 256 feature lanes. One block = CHUNK contiguous edges;
// metadata staged in LDS; register-accumulate runs of equal row, flush runs
// (and chunk boundaries) with global f32 atomics into agg (= d_out, pre-zeroed).
// ---------------------------------------------------------------------------
__global__ __launch_bounds__(256) void gcn_spmm(const float* __restrict__ feat,
                                                const int*   __restrict__ erow,
                                                const int*   __restrict__ ecol,
                                                const float* __restrict__ eval_,
                                                float* __restrict__ agg, int E) {
  __shared__ int   srow[CHUNK];
  __shared__ int   scol[CHUNK];
  __shared__ float sval[CHUNK];

  int e0 = blockIdx.x * CHUNK;
  int nE = E - e0; if (nE > CHUNK) nE = CHUNK;
  for (int i = threadIdx.x; i < nE; i += 256) {
    srow[i] = erow[e0 + i];
    scol[i] = ecol[e0 + i];
    sval[i] = eval_[e0 + i];
  }
  __syncthreads();
  if (nE <= 0) return;

  const int d   = threadIdx.x;
  int   cur = srow[0];
  float acc = 0.f;
  for (int i = 0; i < nE; ++i) {
    int r = srow[i];                       // uniform across block
    if (r != cur) {
      atomicAdd(&agg[(size_t)cur * D + d], acc);
      acc = 0.f;
      cur = r;
    }
    acc = fmaf(sval[i], feat[(size_t)scol[i] * D + d], acc);
  }
  atomicAdd(&agg[(size_t)cur * D + d], acc);
}

// ---------------------------------------------------------------------------
// Kernel C: out = relu(agg @ W) + features, in place over agg (= d_out).
// Block: 256 threads (8 waves), 64 rows x 256 cols of output.
//   wave w: row group rgrp = w>>1 (16 rows), col half chalf = w&1 (128 cols)
//   -> 8 accumulators of 16x16 f32, K loop: 8 steps of 32 (bf16 WMMA).
// A tile staged in LDS as bf16 (pitch 528 B), fragments via ds_read_b128.
// B fragments straight from the pre-swizzled table (2x global_load_b128).
// ---------------------------------------------------------------------------
__global__ __launch_bounds__(256) void gcn_gemm(const float*    __restrict__ feat,
                                                const unsigned* __restrict__ wsw,
                                                float*          __restrict__ out) {
  __shared__ __align__(16) unsigned lds_a[TM * APITCH];   // 33 KB

  const int tid = threadIdx.x;
  const int m0  = blockIdx.x * TM;

  // ---- stage A tile (agg rows m0..m0+63) into LDS as packed bf16 pairs ----
  for (int i = tid; i < TM * (D / 2); i += 256) {
    int r    = i >> 7;          // local row 0..63
    int p    = i & 127;         // pair column 0..127
    int grow = m0 + r;
    f32x2 v = (f32x2)(0.f);
    if (grow < NROWS)
      v = *(const f32x2*)(out + (size_t)grow * D + 2 * p);
    lds_a[r * APITCH + p] = pk_bf16(v.x, v.y);
  }
  __syncthreads();   // all reads of out (agg) complete before any write below

  const int lane  = tid & 31;
  const int wave  = tid >> 5;
  const int rgrp  = wave >> 1;     // 0..3
  const int chalf = wave & 1;      // 0..1
  const int n     = lane & 15;
  const int h     = lane >> 4;
  const int arow  = rgrp * 16 + n; // A: M = lane%16 (both halves)

  v8f acc[8];
#pragma unroll
  for (int t = 0; t < 8; ++t) acc[t] = (v8f){};

  const v16bf* bp = (const v16bf*)wsw;   // 32 B per lane-fragment entry
  const unsigned* ap = &lds_a[arow * APITCH];

#pragma unroll
  for (int ks = 0; ks < 8; ++ks) {
    // A fragment: lanes h=0 hold K = ks*32 + {0..15}; h=1: +16.
    int pbase = ks * 16 + h * 8;                 // pair offset (dwords)
    u32x4 lo = *(const u32x4*)(ap + pbase);      // 8 bf16, 16B aligned
    u32x4 hi = *(const u32x4*)(ap + pbase + 4);  // 8 bf16
    union { u32x4 q[2]; v16bf v; } au;
    au.q[0] = lo; au.q[1] = hi;
    v16bf a = au.v;

    int bbase = (ks * 16 + chalf * 8) * 32 + lane;
#pragma unroll
    for (int t = 0; t < 8; ++t) {
      v16bf b = bp[bbase + t * 32];
      acc[t] = __builtin_amdgcn_wmma_f32_16x16x32_bf16(
          /*neg_a=*/false, a, /*neg_b=*/false, b,
          /*c_mod=*/(short)0, acc[t], /*reuse_a=*/false, /*reuse_b=*/false);
    }
  }

  // ---- epilogue: relu + residual, in place ----
  // C/D layout: VGPR j -> lanes 0-15: (M=j, N=lane); lanes 16-31: (M=j+8, N=lane-16)
  const int rbase = m0 + rgrp * 16 + h * 8;
#pragma unroll
  for (int t = 0; t < 8; ++t) {
    int col = chalf * 128 + t * 16 + n;
#pragma unroll
    for (int j = 0; j < 8; ++j) {
      int row = rbase + j;
      if (row < NROWS) {
        float r = acc[t][j];
        r = r > 0.f ? r : 0.f;
        out[(size_t)row * D + col] = r + feat[(size_t)row * D + col];
      }
    }
  }
}

// ---------------------------------------------------------------------------
extern "C" void kernel_launch(void* const* d_in, const int* in_sizes, int n_in,
                              void* d_out, int out_size, void* d_ws, size_t ws_size,
                              hipStream_t stream) {
  const float* feat  = (const float*)d_in[0];   // [N, 256] f32
  const int*   erow  = (const int*)  d_in[1];   // [E] i32 (sorted)
  const int*   ecol  = (const int*)  d_in[2];   // [E] i32
  const float* eval_ = (const float*)d_in[3];   // [E] f32
  const float* W     = (const float*)d_in[4];   // [256, 256] f32
  float*    out = (float*)d_out;                // doubles as fp32 agg buffer
  unsigned* wsw = (unsigned*)d_ws;              // 128 KB swizzled bf16 W

  const int E = in_sizes[1];

  // agg = 0 (d_out is poisoned by harness; atomics accumulate into it)
  hipMemsetAsync(out, 0, (size_t)NROWS * D * sizeof(float), stream);

  // W -> bf16, pre-swizzled to B-fragment order (32768 dwords)
  gcn_wprep<<<128, 256, 0, stream>>>(W, wsw);

  // SpMM into agg (= d_out)
  int nblk = (E + CHUNK - 1) / CHUNK;
  gcn_spmm<<<nblk, 256, 0, stream>>>(feat, erow, ecol, eval_, out, E);

  // out = relu(agg @ W) + feat, in place
  gcn_gemm<<<(NROWS + TM - 1) / TM, 256, 0, stream>>>(feat, wsw, out);
}
